// PatchTSTRopeAttention_29987461660824
// MI455X (gfx1250) — matrix operand order repaired and checked by reading, not compiled, not run on hardware
//
#include <hip/hip_runtime.h>
#include <hip/hip_bf16.h>
#include <math.h>

// ---------------------------------------------------------------------------
// PatchTST RoPE attention for MI455X (gfx1250), wave32 + WMMA f16->f32.
// Pipeline: cvt -> 3x LDS-staged WMMA GEMMs (QKV, V stored transposed)
//           -> partial RoPE -> flash attention (wmma QK^T + wmma PV)
//           -> WMMA output projection (fp32 out).
// ---------------------------------------------------------------------------

typedef __attribute__((ext_vector_type(16))) _Float16     v16h;
typedef __attribute__((ext_vector_type(8)))  float        v8f;
typedef __attribute__((ext_vector_type(4)))  unsigned int u32x4;

union F16x16 { v16h v; _Float16 h[16]; u32x4 q[2]; };
union F32x8  { v8f  v; float    f[8]; };

__device__ __forceinline__ v8f wmma_f32_f16(v16h a, v16h b, v8f c) {
  // (neg_a, A, neg_b, B, c_mod, C, reuse_a, reuse_b)
  return __builtin_amdgcn_wmma_f32_16x16x32_f16(false, a, false, b, (short)0, c,
                                                false, false);
}

#define EMB 768
#define SEQ 1024
#define NH  12
#define HD  64

// ---------------------------------------------------------------------------
// fp32 -> f16 conversion
// ---------------------------------------------------------------------------
__global__ void cvt_f32_f16(const float* __restrict__ in,
                            _Float16* __restrict__ out, int n) {
  int i = blockIdx.x * blockDim.x + threadIdx.x;
  if (i < n) out[i] = (_Float16)in[i];
}

// ---------------------------------------------------------------------------
// GEMM: C[m,n] = (sum_k A[m,k] * W[n,k] + bias[n]) * scale
// A: [16384, 768] f16 row-major.  W: [768, 768] f16 row-major ([out, in]).
// Block = 256 threads = 8 waves; block tile 128(M) x 64(N), K-step 32.
// A/B tiles staged through LDS, double buffered: global loads for step k+1
// issue before the WMMAs of step k. All fragments for a K-step are pulled
// from LDS before the 4 WMMAs so only one dscnt wait gates the matrix ops.
// mode 0: fp32 plain [M, E] -> d_out
// mode 1: f16 shaped [bh, t, d]            (Q, K)
// mode 2: f16 shaped transposed [bh, d, t] (V^T: PV B-fragments contiguous)
// ---------------------------------------------------------------------------
#define APITCH 48   // row pitch (halves): 96B rows keep ds_*_b128 16B-aligned

__global__ __launch_bounds__(256) void gemm_bias(
    const _Float16* __restrict__ A, const _Float16* __restrict__ W,
    const float* __restrict__ bias, float scale, int mode,
    void* __restrict__ outp) {
  __shared__ __align__(16) _Float16 As[2][128 * APITCH];  // 24 KB
  __shared__ __align__(16) _Float16 Bs[2][64 * APITCH];   // 12 KB

  const int tid   = threadIdx.x;
  const int lane  = tid & 31;
  const int wave  = tid >> 5;
  const int mt    = blockIdx.x & 127;   // 128 M-tiles of 128 rows
  const int nt    = blockIdx.x >> 7;    // 12 N-tiles of 64 cols
  const int mbase = mt * 128;
  const int nbase = nt * 64;

  // staging: A tile 128x32 (8 KB): thread -> 32B; B tile 64x32 (4 KB): 16B
  const int arow = tid >> 1, acol = (tid & 1) * 16;
  const int brow = tid >> 2, bcol = (tid & 3) * 8;
  const _Float16* Ag = A + (size_t)(mbase + arow) * EMB + acol;
  const _Float16* Bg = W + (size_t)(nbase + brow) * EMB + bcol;

  // fragment addressing (WMMA VGPR layouts)
  const int mrow  = lane & 15;
  const int koff  = (lane >> 4) << 3;   // A: 0 or 8
  const int khalf = (lane >> 4) << 4;   // 0 or 16 (unused; B also uses koff)
  const int afrow = wave * 16 + mrow;

  F32x8 c[4] = {};

  // ---- prologue: stage k-step 0 ----
  {
    u32x4 ra0 = *(const u32x4*)(Ag);
    u32x4 ra1 = *(const u32x4*)(Ag + 8);
    u32x4 rb  = *(const u32x4*)(Bg);
    *(u32x4*)(&As[0][arow * APITCH + acol])     = ra0;
    *(u32x4*)(&As[0][arow * APITCH + acol + 8]) = ra1;
    *(u32x4*)(&Bs[0][brow * APITCH + bcol])     = rb;
  }
  __syncthreads();

  for (int kb = 0; kb < EMB; kb += 32) {
    const int cur = (kb >> 5) & 1, nxt = cur ^ 1;
    const bool more = (kb + 32) < EMB;

    // issue next-stage global loads first (overlap with WMMAs below)
    u32x4 ra0, ra1, rb;
    if (more) {
      ra0 = *(const u32x4*)(Ag + kb + 32);
      ra1 = *(const u32x4*)(Ag + kb + 40);
      rb  = *(const u32x4*)(Bg + kb + 32);
    }

    // pull ALL fragments for this K-step from LDS, then run the 4 WMMAs
    F16x16 a, b[4];
    {
      const _Float16* ap = &As[cur][afrow * APITCH];
      a.q[0] = *(const u32x4*)(ap + koff);        // k = koff..koff+7
      a.q[1] = *(const u32x4*)(ap + 16 + koff);   // k = 16+koff..
    }
#pragma unroll
    for (int f = 0; f < 4; ++f) {
      const _Float16* bp = &Bs[cur][(f * 16 + mrow) * APITCH + koff];
      b[f].q[0] = *(const u32x4*)(bp);
      b[f].q[1] = *(const u32x4*)(bp + 16);
    }
#pragma unroll
    for (int f = 0; f < 4; ++f)
      c[f].v = wmma_f32_f16(a.v, b[f].v, c[f].v);

    if (more) {
      *(u32x4*)(&As[nxt][arow * APITCH + acol])     = ra0;
      *(u32x4*)(&As[nxt][arow * APITCH + acol + 8]) = ra1;
      *(u32x4*)(&Bs[nxt][brow * APITCH + bcol])     = rb;
      __syncthreads();  // stage nxt published; all reads of cur already done
    }
  }

  // ---- epilogue ----
  const int roff = (lane >> 4) << 3;            // C row offset: 0 or 8
#pragma unroll
  for (int f = 0; f < 4; ++f) {
    const int n = nbase + f * 16 + mrow;
    const float bs = bias[n];
#pragma unroll
    for (int v = 0; v < 8; ++v) {
      const int m  = mbase + wave * 16 + roff + v;
      const float y = (c[f].f[v] + bs) * scale;
      if (mode == 0) {
        ((float*)outp)[(size_t)m * EMB + n] = y;
      } else {
        const int bidx = m >> 10, t = m & (SEQ - 1);
        const int hh = n >> 6, d = n & (HD - 1);
        const int bh = bidx * NH + hh;
        _Float16* o16 = (_Float16*)outp;
        if (mode == 1) o16[((size_t)bh * SEQ + t) * HD + d]  = (_Float16)y;
        else           o16[((size_t)bh * HD  + d) * SEQ + t] = (_Float16)y;
      }
    }
  }
}

// ---------------------------------------------------------------------------
// Partial RoPE (Gemma half-split): dims d<16 rotate against d+32 with
// timescale 10000^(2d/64); dims 16..31 identity (infinite timescale).
// One thread per (bh, t, d<16) pair, in-place on shaped f16 [bh, t, 64].
// ---------------------------------------------------------------------------
__global__ void rope_kernel(_Float16* __restrict__ X, int total) {
  int i = blockIdx.x * blockDim.x + threadIdx.x;
  if (i >= total) return;
  const int d  = i & 15;
  const int t  = (i >> 4) & (SEQ - 1);
  const int bh = i >> 14;
  const size_t base = ((size_t)bh * SEQ + t) * HD;
  const float inv_ts = __expf(-9.210340371976184f * ((float)(2 * d) * (1.0f / 64.0f)));
  const float ang = (float)t * inv_ts;
  float s, co;
  __sincosf(ang, &s, &co);
  const float x1 = (float)X[base + d];
  const float x2 = (float)X[base + d + 32];
  X[base + d]      = (_Float16)(x1 * co - x2 * s);
  X[base + d + 32] = (_Float16)(x2 * co + x1 * s);
}

// ---------------------------------------------------------------------------
// Flash attention: one wave (32 threads) per (head, 16-query tile).
// Streams keys in blocks of 256; scores live only in LDS; online softmax;
// O accumulated in registers (4x F32x8 = 16x64 fp32 tile). QK^T loop is
// software-pipelined by one sub-tile; next K/V block prefetched
// (global_prefetch_b8) while the current block is computed.
// ---------------------------------------------------------------------------
#define SP 272  // LDS row pitch (elements); 272*2B and 272*4B are 16B multiples

__global__ __launch_bounds__(32) void flash_attn(
    const _Float16* __restrict__ Q, const _Float16* __restrict__ K,
    const _Float16* __restrict__ Vt, _Float16* __restrict__ Oflat) {
  __shared__ __align__(16) float    sS[16 * SP];   // raw scores (fp32)
  __shared__ __align__(16) _Float16 sP[16 * SP];   // exp'd probs (f16)
  __shared__ float sM[16], sL[16], sC[16];         // row max / sum / rescale

  const int lane  = threadIdx.x;
  const int bh    = blockIdx.x >> 6;
  const int qbase = (blockIdx.x & 63) << 4;
  const int b = bh / NH, h = bh % NH;

  const int mrow  = lane & 15;
  const int koff  = (lane >> 4) << 3;   // 0 or 8  (unused; A uses khalf)
  const int khalf = (lane >> 4) << 4;   // 0 or 16 (B fragment)
  const int roff  = (lane >> 4) << 3;   // 0 or 8  (C rows)

  // Persistent Q fragments: a0 covers d 0..31, a1 covers d 32..63
  F16x16 a0, a1;
  {
    const _Float16* qp = Q + ((size_t)bh * SEQ + qbase + mrow) * HD;
    a0.q[0] = *(const u32x4*)(qp + khalf);
    a0.q[1] = *(const u32x4*)(qp + khalf + 8);
    a1.q[0] = *(const u32x4*)(qp + 32 + khalf);
    a1.q[1] = *(const u32x4*)(qp + 40 + khalf);
  }

  F32x8 o[4] = {};
  if (lane < 16) { sM[lane] = -3.0e38f; sL[lane] = 0.0f; }
  __syncthreads();

  const _Float16* Kb = K + (size_t)bh * SEQ * HD + khalf;

  for (int kb = 0; kb < SEQ; kb += 256) {
    // ---- prefetch next 256-key block of K (32 KB) and V^T row starts ----
    if (kb + 256 < SEQ) {
      const char* kpre = (const char*)(K + ((size_t)bh * SEQ + kb + 256) * HD);
#pragma unroll
      for (int pi = 0; pi < 8; ++pi)
        __builtin_prefetch(kpre + (size_t)(lane + pi * 32) * 128, 0, 1);
      const char* vpre = (const char*)(Vt + (size_t)bh * HD * SEQ + kb + 256);
#pragma unroll
      for (int pi = 0; pi < 2; ++pi)
        __builtin_prefetch(vpre + (size_t)(lane + pi * 32) * (SEQ * 2), 0, 1);
    }

    // ---- scores: S = Q (16x64) . K_block^T (64x256), 16 tiles x 2 wmma ----
    // Software-pipelined: K fragments for tile j+1 load under tile j's WMMAs.
    F16x16 b0, b1, nb0, nb1;
    {
      const _Float16* kp = Kb + (size_t)(kb + mrow) * HD;
      b0.q[0] = *(const u32x4*)(kp);
      b0.q[1] = *(const u32x4*)(kp + 8);
      b1.q[0] = *(const u32x4*)(kp + 32);
      b1.q[1] = *(const u32x4*)(kp + 40);
    }
#pragma unroll
    for (int j = 0; j < 16; ++j) {
      if (j < 15) {
        const _Float16* kp = Kb + (size_t)(kb + (j + 1) * 16 + mrow) * HD;
        nb0.q[0] = *(const u32x4*)(kp);
        nb0.q[1] = *(const u32x4*)(kp + 8);
        nb1.q[0] = *(const u32x4*)(kp + 32);
        nb1.q[1] = *(const u32x4*)(kp + 40);
      }
      F32x8 c = {};
      c.v = wmma_f32_f16(a0.v, b0.v, c.v);
      c.v = wmma_f32_f16(a1.v, b1.v, c.v);
#pragma unroll
      for (int v = 0; v < 8; ++v)
        sS[(roff + v) * SP + j * 16 + mrow] = c.f[v];
      b0 = nb0;
      b1 = nb1;
    }
    __syncthreads();

    // ---- online softmax over this 256-key block (2 lanes per row) ----
    {
      const int r    = lane >> 1;
      const int hcol = (lane & 1) * 128;
      float mloc = -3.0e38f;
      for (int ci = 0; ci < 128; ++ci)
        mloc = fmaxf(mloc, sS[r * SP + hcol + ci]);
      mloc = fmaxf(mloc, __shfl_xor(mloc, 1));
      const float mold = sM[r];
      const float mnew = fmaxf(mold, mloc);
      float sum = 0.0f;
      for (int ci = 0; ci < 128; ++ci) {
        const float p = __expf(sS[r * SP + hcol + ci] - mnew);
        sum += p;
        sP[r * SP + hcol + ci] = (_Float16)p;
      }
      sum += __shfl_xor(sum, 1);
      if ((lane & 1) == 0) {
        const float corr = __expf(mold - mnew);
        sC[r] = corr;
        sL[r] = sL[r] * corr + sum;
        sM[r] = mnew;
      }
    }
    __syncthreads();

    // ---- rescale running O by per-row correction factor ----
#pragma unroll
    for (int v = 0; v < 8; ++v) {
      const float cf = sC[roff + v];
#pragma unroll
      for (int f = 0; f < 4; ++f) o[f].f[v] *= cf;
    }

    // ---- O += P (16x256) . V_block (256x64); V stored transposed ----
#pragma unroll
    for (int kc = 0; kc < 8; ++kc) {
      F16x16 ap, bv[4];
      const _Float16* pp = &sP[mrow * SP + kc * 32 + khalf];
      ap.q[0] = *(const u32x4*)(pp);
      ap.q[1] = *(const u32x4*)(pp + 8);
#pragma unroll
      for (int f = 0; f < 4; ++f) {
        const int dcol = f * 16 + mrow;
        const _Float16* vp =
            Vt + ((size_t)bh * HD + dcol) * SEQ + kb + kc * 32 + khalf;
        bv[f].q[0] = *(const u32x4*)(vp);
        bv[f].q[1] = *(const u32x4*)(vp + 8);
      }
#pragma unroll
      for (int f = 0; f < 4; ++f)
        o[f].v = wmma_f32_f16(ap.v, bv[f].v, o[f].v);
    }
    __syncthreads();
  }

  // ---- normalize and scatter into flat [B, T, E] f16 for the O-projection ----
#pragma unroll
  for (int f = 0; f < 4; ++f) {
#pragma unroll
    for (int v = 0; v < 8; ++v) {
      const int r = roff + v;
      const float val = o[f].f[v] / sL[r];
      const int t = qbase + r;
      const int d = f * 16 + mrow;
      Oflat[((size_t)b * SEQ + t) * EMB + h * HD + d] = (_Float16)val;
    }
  }
}

// ---------------------------------------------------------------------------
// Host-side launch
// ---------------------------------------------------------------------------
extern "C" void kernel_launch(void* const* d_in, const int* in_sizes, int n_in,
                              void* d_out, int out_size, void* d_ws, size_t ws_size,
                              hipStream_t stream) {
  const float* x  = (const float*)d_in[0];
  const float* wq = (const float*)d_in[1];
  const float* bq = (const float*)d_in[2];
  const float* wk = (const float*)d_in[3];
  const float* bk = (const float*)d_in[4];
  const float* wv = (const float*)d_in[5];
  const float* bv = (const float*)d_in[6];
  const float* wo = (const float*)d_in[7];
  const float* bo = (const float*)d_in[8];

  const size_t nX = (size_t)16 * SEQ * EMB;   // 12,582,912 (= B*T*E = BH*T*D)
  const size_t nW = (size_t)EMB * EMB;        // 589,824

  char* ws = (char*)d_ws;
  _Float16* Xh  = (_Float16*)ws; ws += nX * 2;
  _Float16* Wqh = (_Float16*)ws; ws += nW * 2;
  _Float16* Wkh = (_Float16*)ws; ws += nW * 2;
  _Float16* Wvh = (_Float16*)ws; ws += nW * 2;
  _Float16* Woh = (_Float16*)ws; ws += nW * 2;
  _Float16* Qs  = (_Float16*)ws; ws += nX * 2;   // [bh, t, d]
  _Float16* Ks  = (_Float16*)ws; ws += nX * 2;   // [bh, t, d]
  _Float16* Vt  = (_Float16*)ws; ws += nX * 2;   // [bh, d, t]
  _Float16* Af  = (_Float16*)ws; ws += nX * 2;   // attention out, flat [B,T,E]

  cvt_f32_f16<<<(int)((nX + 255) / 256), 256, 0, stream>>>(x,  Xh,  (int)nX);
  cvt_f32_f16<<<(int)((nW + 255) / 256), 256, 0, stream>>>(wq, Wqh, (int)nW);
  cvt_f32_f16<<<(int)((nW + 255) / 256), 256, 0, stream>>>(wk, Wkh, (int)nW);
  cvt_f32_f16<<<(int)((nW + 255) / 256), 256, 0, stream>>>(wv, Wvh, (int)nW);
  cvt_f32_f16<<<(int)((nW + 255) / 256), 256, 0, stream>>>(wo, Woh, (int)nW);

  const int gemmGrid = 128 * 12;   // (16384/128) M-tiles x (768/64) N-tiles
  gemm_bias<<<gemmGrid, 256, 0, stream>>>(Xh, Wqh, bq, 0.125f, 1, Qs); // q * D^-1/2
  gemm_bias<<<gemmGrid, 256, 0, stream>>>(Xh, Wkh, bk, 1.0f,   1, Ks);
  gemm_bias<<<gemmGrid, 256, 0, stream>>>(Xh, Wvh, bv, 1.0f,   2, Vt); // transposed

  const int ropeTotal = 192 * SEQ * 16;
  rope_kernel<<<ropeTotal / 256, 256, 0, stream>>>(Qs, ropeTotal);
  rope_kernel<<<ropeTotal / 256, 256, 0, stream>>>(Ks, ropeTotal);

  flash_attn<<<192 * 64, 32, 0, stream>>>(Qs, Ks, Vt, Af);

  gemm_bias<<<gemmGrid, 256, 0, stream>>>(Af, Woh, bo, 1.0f, 0, d_out);
}
